// DuelingQNet_54168127537324
// MI455X (gfx1250) — matrix operand order repaired and checked by reading, not compiled, not run on hardware
//
#include <hip/hip_runtime.h>

typedef __attribute__((ext_vector_type(16))) _Float16 v16h;
typedef __attribute__((ext_vector_type(8)))  _Float16 v8h;
typedef __attribute__((ext_vector_type(4)))  _Float16 v4h;
typedef __attribute__((ext_vector_type(8)))  float    v8f;
typedef __attribute__((ext_vector_type(4)))  float    v4f;
typedef __attribute__((ext_vector_type(4)))  unsigned int u32x4;
typedef __attribute__((ext_vector_type(4)))  int      i32x4;
typedef __attribute__((ext_vector_type(8)))  int      i32x8;

#if defined(__has_builtin)
#if __has_builtin(__builtin_amdgcn_tensor_load_to_lds)
#define HAS_TDM 1
#endif
#endif

static constexpr int kWaves = 4;    // waves per block; each wave owns 32 batch rows
static constexpr int kD     = 128;
static constexpr int kBlk   = 128;  // threads per block
static constexpr int kChunk = 16384; // staged weight chunk (halves) = 32KB

// ---------------- WMMA fragment helpers (CDNA5 16x16x32 f16) ----------------

// A fragment (16 rows x 32 K) from row-major f16, leading dim ld.
static __device__ __forceinline__ v16h a_frag(const _Float16* base, int ld, int lane, int kw) {
  const int row = lane & 15;
  const int kb  = kw + ((lane >> 4) << 4);
  const _Float16* p = base + row * ld + kb;
  v8h lo = *(const v8h*)(p);
  v8h hi = *(const v8h*)(p + 8);
  v16h a;
#pragma unroll
  for (int i = 0; i < 8; ++i) { a[i] = lo[i]; a[i + 8] = hi[i]; }
  return a;
}

#define WMMA_F16(A, B, C) \
  __builtin_amdgcn_wmma_f32_16x16x32_f16(false, (A), false, (B), (short)0, (C), false, false)

// Dual-tile GEMM: B fragments preloaded in groups of 4 (one dscnt wait per 8 WMMAs),
// each B fragment feeds two independent accumulation chains (2x reuse, 2x ILP).
template <int KT>
static __device__ __forceinline__ void gemm_tile2(const v16h* Alo, const v16h* Ahi,
                                                  const _Float16* Wt, int nt, int lane,
                                                  v8f& clo, v8f& chi) {
  const int n  = (lane & 15) + nt * 16;
  const int kb = (lane >> 4) << 4;
  const _Float16* wp = Wt + (size_t)n * (KT * 32) + kb;
#pragma unroll
  for (int g = 0; g < KT; g += 4) {
    v16h b0 = *(const v16h*)(wp + (g + 0) * 32);
    v16h b1 = *(const v16h*)(wp + (g + 1) * 32);
    v16h b2 = *(const v16h*)(wp + (g + 2) * 32);
    v16h b3 = *(const v16h*)(wp + (g + 3) * 32);
    clo = WMMA_F16(Alo[g + 0], b0, clo);  chi = WMMA_F16(Ahi[g + 0], b0, chi);
    clo = WMMA_F16(Alo[g + 1], b1, clo);  chi = WMMA_F16(Ahi[g + 1], b1, chi);
    clo = WMMA_F16(Alo[g + 2], b2, clo);  chi = WMMA_F16(Ahi[g + 2], b2, chi);
    clo = WMMA_F16(Alo[g + 3], b3, clo);  chi = WMMA_F16(Ahi[g + 3], b3, chi);
  }
}

// D tile (f32 C layout: lane n = col, vgpr r = row (lane>=16)*8+r) -> f16 row-major LDS.
static __device__ __forceinline__ void store_tile_f16(_Float16* dst, int ld, int lane,
                                                      int nt, v8f c, const float* bias,
                                                      bool relu) {
  const int col = (lane & 15) + nt * 16;
  const int rb  = (lane >> 4) << 3;
  const float b = bias[col];
#pragma unroll
  for (int r = 0; r < 8; ++r) {
    float v = c[r] + b;
    if (relu) v = fmaxf(v, 0.f);
    dst[(rb + r) * ld + col] = (_Float16)v;
  }
}

static __device__ __forceinline__ void store_tile_f32(float* dst /*[16][16]*/, int lane,
                                                      v8f c) {
  const int col = lane & 15;
  const int rb  = (lane >> 4) << 3;
#pragma unroll
  for (int r = 0; r < 8; ++r) dst[(rb + r) * 16 + col] = c[r];
}

static __device__ __forceinline__ void store_tile_f32_bias(float* dst, int lane, v8f c,
                                                           const float* bias, int nbias) {
  const int col = lane & 15;
  const int rb  = (lane >> 4) << 3;
  const float b = (col < nbias) ? bias[col] : 0.f;
#pragma unroll
  for (int r = 0; r < 8; ++r) dst[(rb + r) * 16 + col] = c[r] + b;
}

static __device__ __forceinline__ v8h splat8(_Float16 x) {
  v8h r = {x, x, x, x, x, x, x, x};
  return r;
}

// ---------------- TDM staging: global f16 arena -> LDS ----------------
// D# per CDNA5 ISA: count=1, type=2 (image), data_size=1 (2B),
// 2D tile == tensor: rows of 2048 halves, stride 2048. Groups 2/3 zero.
static __device__ __forceinline__ void tdm_issue(const _Float16* gsrc, unsigned lds_off,
                                                 int halves) {
#if defined(HAS_TDM)
  unsigned long long ga = (unsigned long long)gsrc;
  const unsigned rows = (unsigned)(halves >> 11);       // halves / 2048
  u32x4 g0 = { 1u,                                      // count=1, user mode
               lds_off,                                  // lds_addr [63:32]
               (unsigned)ga,                             // global_addr low
               (unsigned)((ga >> 32) & 0x01FFFFFFu) | 0x80000000u }; // addr[56:32]|type=2
  i32x8 g1 = { (int)(1u << 16),                         // data_size=1 (2 bytes)
               (int)(2048u << 16),                      // tensor_dim0 low16 @ [31:16]
               (int)(rows << 16),                       // td0 hi=0 | tensor_dim1 low16
               (int)(2048u << 16),                      // td1 hi=0 | tile_dim0
               (int)rows,                               // tile_dim1 | tile_dim2=0
               2048,                                    // tensor_dim0_stride low32
               0, 0 };
  i32x4 gz = { 0, 0, 0, 0 };
#if __clang_major__ >= 23
  i32x8 g4 = { 0, 0, 0, 0, 0, 0, 0, 0 };
  __builtin_amdgcn_tensor_load_to_lds(g0, g1, gz, gz, g4, 0);
#else
  __builtin_amdgcn_tensor_load_to_lds(g0, g1, gz, gz, 0);
#endif
#else
  (void)gsrc; (void)lds_off; (void)halves;
#endif
}

// Pipelined stage step: kick stage i+1 into its buffer, then make stage i resident.
// WN = tensorcnt to wait for (1 while pipelining, 0 on the final step).
template <int WN>
static __device__ __forceinline__ void pipe_step(int tid, const _Float16* next_src,
                                                 int next_halves, _Float16* next_dst) {
  __syncthreads();                 // all waves done reading next_dst's old contents
#if defined(HAS_TDM)
  if (tid == 0) {
    if (next_src) tdm_issue(next_src, (unsigned)(unsigned long long)next_dst, next_halves);
    __builtin_amdgcn_s_wait_tensorcnt(WN);   // in-order TDM: current stage now resident
  }
#else
  if (next_src)
    for (int i = tid * 8; i < next_halves; i += kBlk * 8)
      *(v8h*)(next_dst + i) = *(const v8h*)(next_src + i);
#endif
  __syncthreads();                 // staged data visible to all waves
}

// ---------------- weight transpose + fp32->f16 convert ----------------
// dst[n][k] = (n < N) ? src[k][n] : 0  for n < Npad (pads narrow heads to N=16)
__global__ void tconv_kernel(const float* __restrict__ src, _Float16* __restrict__ dst,
                             int K, int N, int Npad) {
  int t = blockIdx.x * blockDim.x + threadIdx.x;
  int total = K * Npad;
  if (t >= total) return;
  int n = t / K, k = t - n * K;
  float v = (n < N) ? src[(size_t)k * N + n] : 0.f;
  dst[(size_t)n * K + k] = (_Float16)v;
}

// ---------------- fused SoftModule dueling-Q forward ----------------

__global__ __launch_bounds__(kBlk)
void qnet_fused(const float* __restrict__ o, const float* __restrict__ zmap,
                const float* __restrict__ cprob,
                const float* __restrict__ Wz, const float* __restrict__ bb,
                const float* __restrict__ bz, const float* __restrict__ bm,
                const float* __restrict__ Wv, const float* __restrict__ bv,
                const float* __restrict__ ba1, const float* __restrict__ ba2,
                const _Float16* __restrict__ tWb,  const _Float16* __restrict__ tWm,
                const _Float16* __restrict__ tWg,  const _Float16* __restrict__ tWgl,
                const _Float16* __restrict__ tWa1, /* tWa2 = tWa1 + 16384 */
                float* __restrict__ out) {
  __shared__ __align__(32) _Float16 xbuf[kWaves][4][32][kD];  // module acts (+ o stage)
  __shared__ __align__(32) _Float16 fbuf[kWaves][32][kD];     // f0, later feature
  __shared__ __align__(32) _Float16 gbuf[kWaves][32][kD];     // g, later hidden h
  __shared__ __align__(32) _Float16 pbuf[kWaves][32][16];     // routing probs (f16)
  __shared__ __align__(32) float    lbuf[kWaves][32][16];     // logits / adv scratch
  __shared__ __align__(256) _Float16 wbuf[2][kChunk];         // 2x32KB staged weights
  __shared__ __align__(32) float    wzs[10 * kD];
  __shared__ __align__(32) float    bzs[kD];

  const int tid  = threadIdx.x;
  const int w    = tid >> 5;
  const int lane = tid & 31;
  const int r0   = (blockIdx.x * kWaves + w) * 32;   // batch row base for this wave

  // Stage tiny routing-input weights once per block.
  for (int i = tid; i < 10 * kD; i += kBlk) wzs[i] = Wz[i];
  for (int i = tid; i < kD; i += kBlk)      bzs[i] = bz[i];

  __builtin_prefetch(&o[(size_t)r0 * 256], 0, 3);    // global_prefetch_b8

  // Kick stage S0 (Wb rows 0..63); the o-tile staging below overlaps the DMA.
  __syncthreads();
#if defined(HAS_TDM)
  if (tid == 0) tdm_issue(tWb, (unsigned)(unsigned long long)&wbuf[0][0], kChunk);
#else
  for (int i = tid * 8; i < kChunk; i += kBlk * 8)
    *(v8h*)(&wbuf[0][0] + i) = *(const v8h*)(tWb + i);
#endif

  // ---- stage this wave's o tile (32x256 -> f16, coalesced) ----
  _Float16* ost = &xbuf[w][0][0][0];                 // viewed as [32][256]
#pragma unroll
  for (int it = 0; it < 64; ++it) {
    int e4  = ((it * 32 + lane) << 2);
    int row = e4 >> 8, kc = e4 & 255;
    v4f v = *(const v4f*)&o[(size_t)(r0 + row) * 256 + kc];
    v4h h = {(_Float16)v[0], (_Float16)v[1], (_Float16)v[2], (_Float16)v[3]};
    *(v4h*)&ost[row * 256 + kc] = h;
  }

  // ---- f0 = relu(o @ Wb + bb): dual 16-row tiles, Wb staged in 2 n-chunks ----
  {
    v16h Alo[8], Ahi[8];
#pragma unroll
    for (int kt = 0; kt < 8; ++kt) {
      Alo[kt] = a_frag(ost,            256, lane, kt * 32);
      Ahi[kt] = a_frag(ost + 16 * 256, 256, lane, kt * 32);
    }
    // C0: nt 0..3 from wbuf[0]; kick S1 = Wb rows 64..127
    pipe_step<1>(tid, tWb + kChunk, kChunk, &wbuf[1][0]);
#pragma unroll 2
    for (int nt = 0; nt < 4; ++nt) {
      v8f clo = {0.f,0.f,0.f,0.f,0.f,0.f,0.f,0.f};
      v8f chi = {0.f,0.f,0.f,0.f,0.f,0.f,0.f,0.f};
      gemm_tile2<8>(Alo, Ahi, &wbuf[0][0], nt, lane, clo, chi);
      store_tile_f16(&fbuf[w][0][0],  kD, lane, nt, clo, bb, true);
      store_tile_f16(&fbuf[w][16][0], kD, lane, nt, chi, bb, true);
    }
    // C1: nt 4..7 from wbuf[1]; kick S2 = Wm[0][0]
    pipe_step<1>(tid, tWm, kChunk, &wbuf[0][0]);
#pragma unroll 2
    for (int nt = 4; nt < 8; ++nt) {
      v8f clo = {0.f,0.f,0.f,0.f,0.f,0.f,0.f,0.f};
      v8f chi = {0.f,0.f,0.f,0.f,0.f,0.f,0.f,0.f};
      gemm_tile2<8>(Alo, Ahi, &wbuf[1][0], nt - 4, lane, clo, chi);
      store_tile_f16(&fbuf[w][0][0],  kD, lane, nt, clo, bb, true);
      store_tile_f16(&fbuf[w][16][0], kD, lane, nt, chi, bb, true);
    }
  }

  // ---- g = relu(z_in @ Wz + bz) * f0  (K=10 VALU path; lane == row) ----
  {
    const int row = lane;
    float z[10];
#pragma unroll
    for (int k = 0; k < 9; ++k) z[k] = zmap[(size_t)(r0 + row) * 9 + k];
    z[9] = cprob[r0 + row];
    for (int d = 0; d < kD; ++d) {
      float s = bzs[d];
#pragma unroll
      for (int k = 0; k < 10; ++k) s += z[k] * wzs[k * kD + d];
      s = fmaxf(s, 0.f);
      gbuf[w][row][d] = (_Float16)(s * (float)fbuf[w][row][d]);
    }
  }

  // ---- layers 0..3: (gate -> softmax -> mix for l>=1) + 4 module GEMMs ----
  // Stage index: s = 2 + l*4 + m (module weights), 18 = Wa1, 19 = Wa2.
  for (int l = 0; l < 4; ++l) {
    if (l > 0) {
      {   // logits (16x16 per tile), gate head from global (tiny, L1-resident)
        v16h Aglo[4], Aghi[4];
#pragma unroll
        for (int kt = 0; kt < 4; ++kt) {
          Aglo[kt] = a_frag(&gbuf[w][0][0],  kD, lane, kt * 32);
          Aghi[kt] = a_frag(&gbuf[w][16][0], kD, lane, kt * 32);
        }
        v8f clo = {0.f,0.f,0.f,0.f,0.f,0.f,0.f,0.f};
        v8f chi = {0.f,0.f,0.f,0.f,0.f,0.f,0.f,0.f};
        gemm_tile2<4>(Aglo, Aghi, tWg + (size_t)(l - 1) * 16 * kD, 0, lane, clo, chi);
        store_tile_f32(&lbuf[w][0][0],  lane, clo);
        store_tile_f32(&lbuf[w][16][0], lane, chi);
      }
      {   // softmax over incoming module i, per (row=lane, j); logit index = i*4+j
        const int row = lane;
#pragma unroll
        for (int j = 0; j < 4; ++j) {
          float v0 = lbuf[w][row][0 + j], v1 = lbuf[w][row][4 + j];
          float v2 = lbuf[w][row][8 + j], v3 = lbuf[w][row][12 + j];
          float mx = fmaxf(fmaxf(v0, v1), fmaxf(v2, v3));
          float e0 = __expf(v0 - mx), e1 = __expf(v1 - mx);
          float e2 = __expf(v2 - mx), e3 = __expf(v3 - mx);
          float inv = 1.f / (e0 + e1 + e2 + e3);
          pbuf[w][row][0 + j]  = (_Float16)(e0 * inv);
          pbuf[w][row][4 + j]  = (_Float16)(e1 * inv);
          pbuf[w][row][8 + j]  = (_Float16)(e2 * inv);
          pbuf[w][row][12 + j] = (_Float16)(e3 * inv);
        }
      }
      // in-place mix: xin[j] = sum_i p[i][j] * x[i]  (packed f16)
#pragma unroll
      for (int t = 0; t < 16; ++t) {
        int u = t * 32 + lane;
        int row = u >> 4;
        int ch  = (u & 15) << 3;
        v16h pr = *(const v16h*)&pbuf[w][row][0];
        v8h x0 = *(const v8h*)&xbuf[w][0][row][ch];
        v8h x1 = *(const v8h*)&xbuf[w][1][row][ch];
        v8h x2 = *(const v8h*)&xbuf[w][2][row][ch];
        v8h x3 = *(const v8h*)&xbuf[w][3][row][ch];
#pragma unroll
        for (int j = 0; j < 4; ++j) {
          v8h acc = x0 * splat8(pr[j]) + x1 * splat8(pr[4 + j]) +
                    x2 * splat8(pr[8 + j]) + x3 * splat8(pr[12 + j]);
          *(v8h*)&xbuf[w][j][row][ch] = acc;
        }
      }
    }
    // 4 module GEMMs, staged one module per 32KB chunk, pipelined
    for (int m = 0; m < 4; ++m) {
      const int s  = 2 + l * 4 + m;          // current stage index
      const int sn = s + 1;                  // next stage to kick
      const _Float16* nsrc;
      int nhalves;
      if (sn <= 17)      { nsrc = tWm + (size_t)(sn - 2) * kChunk; nhalves = kChunk; }
      else if (sn == 18) { nsrc = tWa1;                            nhalves = kChunk; }
      else               { nsrc = tWa1 + kChunk;                   nhalves = 2048;   }
      pipe_step<1>(tid, nsrc, nhalves, &wbuf[sn & 1][0]);

      const _Float16* Wt = &wbuf[s & 1][0];
      const _Float16* Ab = (l == 0) ? &fbuf[w][0][0] : &xbuf[w][m][0][0];
      v16h Alo[4], Ahi[4];
#pragma unroll
      for (int kt = 0; kt < 4; ++kt) {
        Alo[kt] = a_frag(Ab,            kD, lane, kt * 32);
        Ahi[kt] = a_frag(Ab + 16 * kD,  kD, lane, kt * 32);
      }
      const float* bias = bm + (l * 4 + m) * kD;
#pragma unroll 2
      for (int nt = 0; nt < 8; ++nt) {
        v8f clo = {0.f,0.f,0.f,0.f,0.f,0.f,0.f,0.f};
        v8f chi = {0.f,0.f,0.f,0.f,0.f,0.f,0.f,0.f};
        gemm_tile2<4>(Alo, Ahi, Wt, nt, lane, clo, chi);
        store_tile_f16(&xbuf[w][m][0][0],  kD, lane, nt, clo, bias, true);
        store_tile_f16(&xbuf[w][m][16][0], kD, lane, nt, chi, bias, true);
      }
    }
  }

  // ---- p_last = softmax(g @ Wg_last) (padded N=16, cols 0..3 valid) ----
  {
    v16h Aglo[4], Aghi[4];
#pragma unroll
    for (int kt = 0; kt < 4; ++kt) {
      Aglo[kt] = a_frag(&gbuf[w][0][0],  kD, lane, kt * 32);
      Aghi[kt] = a_frag(&gbuf[w][16][0], kD, lane, kt * 32);
    }
    v8f clo = {0.f,0.f,0.f,0.f,0.f,0.f,0.f,0.f};
    v8f chi = {0.f,0.f,0.f,0.f,0.f,0.f,0.f,0.f};
    gemm_tile2<4>(Aglo, Aghi, tWgl, 0, lane, clo, chi);
    store_tile_f32(&lbuf[w][0][0],  lane, clo);
    store_tile_f32(&lbuf[w][16][0], lane, chi);
    const int row = lane;
    float v0 = lbuf[w][row][0], v1 = lbuf[w][row][1];
    float v2 = lbuf[w][row][2], v3 = lbuf[w][row][3];
    float mx = fmaxf(fmaxf(v0, v1), fmaxf(v2, v3));
    float e0 = __expf(v0 - mx), e1 = __expf(v1 - mx);
    float e2 = __expf(v2 - mx), e3 = __expf(v3 - mx);
    float inv = 1.f / (e0 + e1 + e2 + e3);
    pbuf[w][row][0] = (_Float16)(e0 * inv);
    pbuf[w][row][1] = (_Float16)(e1 * inv);
    pbuf[w][row][2] = (_Float16)(e2 * inv);
    pbuf[w][row][3] = (_Float16)(e3 * inv);
  }

  // ---- feature = sum_m p_last[m] * x[m]  -> fbuf ----
#pragma unroll
  for (int t = 0; t < 16; ++t) {
    int u = t * 32 + lane;
    int row = u >> 4;
    int ch  = (u & 15) << 3;
    _Float16 p0 = pbuf[w][row][0], p1 = pbuf[w][row][1];
    _Float16 p2 = pbuf[w][row][2], p3 = pbuf[w][row][3];
    v8h acc = (*(const v8h*)&xbuf[w][0][row][ch]) * splat8(p0) +
              (*(const v8h*)&xbuf[w][1][row][ch]) * splat8(p1) +
              (*(const v8h*)&xbuf[w][2][row][ch]) * splat8(p2) +
              (*(const v8h*)&xbuf[w][3][row][ch]) * splat8(p3);
    *(v8h*)&fbuf[w][row][ch] = acc;
  }

  // ---- value = feature @ Wv + bv (N=1 VALU dot; lane == row) ----
  float val = bv[0];
  {
    const int row = lane;
    for (int d = 0; d < kD; ++d) val += (float)fbuf[w][row][d] * Wv[d];
  }

  // ---- C18: h = relu(feature @ Wa1 + ba1), Wa1 resident in wbuf[0]; kick Wa2 ----
  pipe_step<1>(tid, tWa1 + kChunk, 2048, &wbuf[1][0]);
  {
    v16h Alo[4], Ahi[4];
#pragma unroll
    for (int kt = 0; kt < 4; ++kt) {
      Alo[kt] = a_frag(&fbuf[w][0][0],  kD, lane, kt * 32);
      Ahi[kt] = a_frag(&fbuf[w][16][0], kD, lane, kt * 32);
    }
#pragma unroll 2
    for (int nt = 0; nt < 8; ++nt) {
      v8f clo = {0.f,0.f,0.f,0.f,0.f,0.f,0.f,0.f};
      v8f chi = {0.f,0.f,0.f,0.f,0.f,0.f,0.f,0.f};
      gemm_tile2<4>(Alo, Ahi, &wbuf[0][0], nt, lane, clo, chi);
      store_tile_f16(&gbuf[w][0][0],  kD, lane, nt, clo, ba1, true);
      store_tile_f16(&gbuf[w][16][0], kD, lane, nt, chi, ba1, true);
    }
  }
  // ---- C19: adv = h @ Wa2 + ba2 (N=8 padded to 16), final stage, wait to 0 ----
  pipe_step<0>(tid, nullptr, 0, nullptr);
  {
    v16h Alo[4], Ahi[4];
#pragma unroll
    for (int kt = 0; kt < 4; ++kt) {
      Alo[kt] = a_frag(&gbuf[w][0][0],  kD, lane, kt * 32);
      Ahi[kt] = a_frag(&gbuf[w][16][0], kD, lane, kt * 32);
    }
    v8f clo = {0.f,0.f,0.f,0.f,0.f,0.f,0.f,0.f};
    v8f chi = {0.f,0.f,0.f,0.f,0.f,0.f,0.f,0.f};
    gemm_tile2<4>(Alo, Ahi, &wbuf[1][0], 0, lane, clo, chi);
    store_tile_f32_bias(&lbuf[w][0][0],  lane, clo, ba2, 8);
    store_tile_f32_bias(&lbuf[w][16][0], lane, chi, ba2, 8);
  }

  // ---- finalize: out = adv - mean(adv) + value (lane == row) ----
  {
    const int row = lane;
    float a[8], s = 0.f;
#pragma unroll
    for (int i = 0; i < 8; ++i) { a[i] = lbuf[w][row][i]; s += a[i]; }
    const float mean = s * 0.125f;
    float* po = out + (size_t)(r0 + row) * 8;
#pragma unroll
    for (int i = 0; i < 8; ++i) po[i] = a[i] - mean + val;
  }
}

// ---------------- host launcher ----------------

extern "C" void kernel_launch(void* const* d_in, const int* in_sizes, int n_in,
                              void* d_out, int out_size, void* d_ws, size_t ws_size,
                              hipStream_t stream) {
  (void)in_sizes; (void)n_in; (void)ws_size;
  const float* o    = (const float*)d_in[0];
  const float* zmap = (const float*)d_in[1];
  const float* cp   = (const float*)d_in[2];
  const float* Wb   = (const float*)d_in[3];
  const float* bb   = (const float*)d_in[4];
  const float* Wz   = (const float*)d_in[5];
  const float* bz   = (const float*)d_in[6];
  const float* Wm   = (const float*)d_in[7];
  const float* bm   = (const float*)d_in[8];
  const float* Wg   = (const float*)d_in[9];
  const float* Wgl  = (const float*)d_in[10];
  const float* Wv   = (const float*)d_in[11];
  const float* bv   = (const float*)d_in[12];
  const float* Wa1  = (const float*)d_in[13];
  const float* ba1  = (const float*)d_in[14];
  const float* Wa2  = (const float*)d_in[15];
  const float* ba2  = (const float*)d_in[16];
  float* out = (float*)d_out;

  // f16 transposed-weight arena in workspace (offsets 4KB-aligned; Wa1|Wa2 contiguous)
  _Float16* ws   = (_Float16*)d_ws;
  _Float16* tWb  = ws;                 // 128 x 256
  _Float16* tWm  = tWb + 32768;        // 16 x (128 x 128)
  _Float16* tWg  = tWm + 262144;       // 3 x (16 x 128)
  _Float16* tWgl = tWg + 6144;         // 16 x 128 (rows 4..15 zero)
  _Float16* tWa1 = tWgl + 2048;        // 128 x 128
  _Float16* tWa2 = tWa1 + 16384;       // 16 x 128 (rows 8..15 zero)

  auto tc = [&](const float* s, _Float16* d, int K, int N, int Np) {
    int tot = K * Np;
    tconv_kernel<<<(tot + 255) / 256, 256, 0, stream>>>(s, d, K, N, Np);
  };
  tc(Wb, tWb, 256, 128, 128);
  for (int i = 0; i < 16; ++i) tc(Wm + (size_t)i * 16384, tWm + (size_t)i * 16384, 128, 128, 128);
  for (int l = 0; l < 3; ++l)  tc(Wg + (size_t)l * 128 * 16, tWg + (size_t)l * 2048, 128, 16, 16);
  tc(Wgl, tWgl, 128, 4, 16);
  tc(Wa1, tWa1, 128, 128, 128);
  tc(Wa2, tWa2, 128, 8, 16);

  const int rows    = out_size / 8;            // batch size
  const int nblocks = rows / (32 * kWaves);    // 65536/128 = 512
  qnet_fused<<<nblocks, kBlk, 0, stream>>>(o, zmap, cp, Wz, bb, bz, bm, Wv, bv, ba1, ba2,
                                           tWb, tWm, tWg, tWgl, tWa1, out);
}
